// SimpleRNN_60266981097547
// MI455X (gfx1250) — matrix operand order repaired and checked by reading, not compiled, not run on hardware
//
#include <hip/hip_runtime.h>
#include <hip/hip_bf16.h>
#include <math.h>

// ---------------------------------------------------------------------------
// SimpleRNN on MI455X (gfx1250): bf16 WMMA recurrence + WMMA output projection
// ---------------------------------------------------------------------------

typedef __bf16 bf16_t;
typedef __attribute__((ext_vector_type(16))) __bf16 v16bf;
typedef __attribute__((ext_vector_type(8)))  float  v8f;

union FragBF {
    v16bf v;
    uint4 q[2];   // 32 bytes
};

#define B_DIM   64
#define S_LEN   512
#define H_DIM   1024
#define V_DIM   128
#define NWG_REC 16

// Branch-free fast tanh: 1 - 2/(1 + exp2(2*log2(e)*x)).
// v_exp_f32 + v_rcp_f32; saturates correctly at +/-1 for large |x|.
__device__ __forceinline__ float fast_tanh(float x) {
    float e = __builtin_amdgcn_exp2f(x * 2.8853900817779268f);
    return 1.0f - 2.0f * __builtin_amdgcn_rcpf(1.0f + e);
}

// ---------------------------------------------------------------------------
// Two-phase grid barrier (count + generation) in global workspace.
// ---------------------------------------------------------------------------
__device__ __forceinline__ void grid_barrier(unsigned* bar, unsigned nwg) {
    __syncthreads();
    __threadfence();                       // release writes of this step
    if (threadIdx.x == 0) {
        volatile unsigned* gen = bar + 1;
        unsigned g = *gen;
        if (atomicAdd(bar, 1u) == nwg - 1u) {
            bar[0] = 0u;
            __threadfence();
            atomicAdd((unsigned*)(bar + 1), 1u);
        } else {
            while (*gen == g) { __builtin_amdgcn_s_sleep(2); }
            __threadfence();               // acquire other blocks' writes
        }
    }
    __syncthreads();
}

// ---------------------------------------------------------------------------
// init: h0 (fp32) -> hbuf slot0 (bf16); reset barrier state
// ---------------------------------------------------------------------------
__global__ __launch_bounds__(256) void rnn_init_kernel(
        const float* __restrict__ h0, bf16_t* __restrict__ hbuf0,
        unsigned* __restrict__ bar) {
    int i = blockIdx.x * 256 + threadIdx.x;
    if (i < B_DIM * H_DIM) hbuf0[i] = (bf16_t)h0[i];
    if (i == 0) { bar[0] = 0u; bar[1] = 0u; }
}

// ---------------------------------------------------------------------------
// transpose fp32 (R x C) -> bf16 (C x R), 32x32 LDS tiles
// ---------------------------------------------------------------------------
__global__ __launch_bounds__(256) void rnn_transpose_bf16_kernel(
        const float* __restrict__ in, bf16_t* __restrict__ out, int R, int C) {
    __shared__ float tile[32][33];
    int x = blockIdx.x * 32 + threadIdx.x;           // input col
    for (int j = threadIdx.y; j < 32; j += 8) {
        int y = blockIdx.y * 32 + j;                 // input row
        tile[j][threadIdx.x] = (y < R && x < C) ? in[(size_t)y * C + x] : 0.0f;
    }
    __syncthreads();
    int ox = blockIdx.y * 32 + threadIdx.x;          // output col (= input row)
    for (int j = threadIdx.y; j < 32; j += 8) {
        int oy = blockIdx.x * 32 + j;                // output row (= input col)
        if (oy < C && ox < R)
            out[(size_t)oy * R + ox] = (bf16_t)tile[threadIdx.x][j];
    }
}

// ---------------------------------------------------------------------------
// Phase 1: persistent recurrence kernel.
//   h_{t+1} = tanh( W_xh[x_t] + h_t @ W_hh + b_h )
// 16 WGs x 256 threads (8 waves). WG g owns output columns [64g, 64g+64).
// LDS: W_hh^T slice [64 n][1024 k] bf16, row stride 1032 (132 KB).
// Wave w: m-tile = w&3 (16 batch rows), two adjacent 16-col N-tiles (w>>2).
// Per step: 64 x v_wmma_f32_16x16x32_bf16 (A fragment reused 2x), then a
// fused epilogue (token/embedding loads hoisted ahead of the K-loop so their
// latency hides under the WMMAs), then one global barrier.
// ---------------------------------------------------------------------------
__global__ __launch_bounds__(256) void rnn_recurrence_kernel(
        const int*    __restrict__ x,       // (B,S)
        const float*  __restrict__ W_xh,    // (V,H) fp32
        const float*  __restrict__ b_h,     // (H)
        const bf16_t* __restrict__ whhT,    // (H n, H k) bf16 = W_hh^T
        bf16_t*       __restrict__ hbuf,    // 2 x (B,H) bf16 double buffer
        bf16_t*       __restrict__ hidden,  // (B,S,H) bf16
        float*        __restrict__ hfinal,  // (B,H) fp32
        unsigned*     __restrict__ bar) {
    extern __shared__ char smem[];
    bf16_t* wlds = reinterpret_cast<bf16_t*>(smem);
    const int LDW = H_DIM + 8;                       // pad: stride 1032 bf16

    const int n0g = blockIdx.x * 64;

    // cooperative copy of W_hh^T rows [n0g, n0g+64) into LDS (uint4 = 8 bf16)
    for (int i = threadIdx.x; i < 64 * (H_DIM / 8); i += 256) {
        int row = i >> 7;                  // / (1024/8)
        int c   = (i & 127) << 3;
        *reinterpret_cast<uint4*>(wlds + row * LDW + c) =
            *reinterpret_cast<const uint4*>(whhT + (size_t)(n0g + row) * H_DIM + c);
    }
    __syncthreads();

    const int wave = threadIdx.x >> 5;
    const int lane = threadIdx.x & 31;
    const int m0   = (wave & 3) * 16;                // batch-row tile
    const int n0l  = (wave >> 2) * 32;               // local col pair base
    const int hi8  = (lane & 16) >> 1;               // 0 or 8 (C-layout rows)
    const int nn0  = n0g + n0l + (lane & 15);        // output col, tile 0
    const int nn1  = nn0 + 16;                       // output col, tile 1
    const float bh0 = b_h[nn0];
    const float bh1 = b_h[nn1];

    // fragment base pointers (A from global h, B from LDS W_hh^T slice)
    const int a_row  = m0 + (lane & 15);
    const int a_koff = (lane & 16);                  // +16 elements for hi half
    const bf16_t* brow0 = wlds + (n0l + (lane & 15)) * LDW + (lane & 16);
    const bf16_t* brow1 = brow0 + 16 * LDW;

    const v8f vzero = {0.f, 0.f, 0.f, 0.f, 0.f, 0.f, 0.f, 0.f};

    for (int t = 0; t < S_LEN; ++t) {
        const bf16_t* hcur = hbuf + (size_t)(t & 1) * (B_DIM * H_DIM);
        bf16_t*       hnxt = hbuf + (size_t)((t + 1) & 1) * (B_DIM * H_DIM);

        // --- hoisted token + embedding gathers: independent of the K-loop,
        //     so their latency overlaps the 64 WMMAs below.
        int tok[8];
        #pragma unroll
        for (int r = 0; r < 8; ++r)
            tok[r] = x[(m0 + hi8 + r) * S_LEN + t];
        float emb0[8], emb1[8];
        #pragma unroll
        for (int r = 0; r < 8; ++r) {
            const float* wrow = W_xh + (size_t)tok[r] * H_DIM;
            emb0[r] = wrow[nn0];
            emb1[r] = wrow[nn1];
        }

        v8f acc0 = vzero, acc1 = vzero;
        const bf16_t* arow = hcur + (size_t)a_row * H_DIM + a_koff;

        #pragma unroll 4
        for (int ks = 0; ks < H_DIM / 32; ++ks) {
            FragBF fa, fb0, fb1;
            const bf16_t* pa = arow + ks * 32;
            fa.q[0] = *reinterpret_cast<const uint4*>(pa);        // k0+16h..+7
            fa.q[1] = *reinterpret_cast<const uint4*>(pa + 8);    // +8 elems
            const bf16_t* pb0 = brow0 + ks * 32;
            fb0.q[0] = *reinterpret_cast<const uint4*>(pb0);      // 32B contig
            fb0.q[1] = *reinterpret_cast<const uint4*>(pb0 + 8);
            const bf16_t* pb1 = brow1 + ks * 32;
            fb1.q[0] = *reinterpret_cast<const uint4*>(pb1);
            fb1.q[1] = *reinterpret_cast<const uint4*>(pb1 + 8);
            acc0 = __builtin_amdgcn_wmma_f32_16x16x32_bf16(
                       false, fa.v, false, fb0.v, (short)0, acc0, false, false);
            acc1 = __builtin_amdgcn_wmma_f32_16x16x32_bf16(
                       false, fa.v, false, fb1.v, (short)0, acc1, false, false);
        }

        // fused epilogue: bias + embedding, fast tanh, bf16 stores
        #pragma unroll
        for (int r = 0; r < 8; ++r) {
            int b = m0 + hi8 + r;
            float hv0 = fast_tanh(acc0[r] + emb0[r] + bh0);
            float hv1 = fast_tanh(acc1[r] + emb1[r] + bh1);
            bf16_t hb0 = (bf16_t)hv0;
            bf16_t hb1 = (bf16_t)hv1;
            hnxt[(size_t)b * H_DIM + nn0] = hb0;
            hnxt[(size_t)b * H_DIM + nn1] = hb1;
            bf16_t* hrow = hidden + ((size_t)b * S_LEN + t) * H_DIM;
            hrow[nn0] = hb0;
            hrow[nn1] = hb1;
            if (t == S_LEN - 1) {
                hfinal[(size_t)b * H_DIM + nn0] = hv0;
                hfinal[(size_t)b * H_DIM + nn1] = hv1;
            }
        }

        grid_barrier(bar, NWG_REC);
    }
}

// ---------------------------------------------------------------------------
// Phase 2: logits = hidden @ W_out + b_out.  (32768 x 1024) @ (1024 x 128)
// 256 WGs x 8 waves; wave computes a 16(M) x 128(N) strip: 8 accumulators,
// A fragment reused across all 8 N-tiles. W_out^T staged fully in LDS.
// ---------------------------------------------------------------------------
__global__ __launch_bounds__(256) void rnn_logits_kernel(
        const bf16_t* __restrict__ hidden,  // (B*S, H) bf16
        const bf16_t* __restrict__ woutT,   // (V, H) bf16 = W_out^T
        const float*  __restrict__ b_out,   // (V)
        float*        __restrict__ logits)  // (B*S, V) fp32
{
    extern __shared__ char smem[];
    bf16_t* blds = reinterpret_cast<bf16_t*>(smem);
    const int LDB = H_DIM + 8;

    // stage W_out^T (128 x 1024 bf16 = 256 KB) into LDS
    for (int i = threadIdx.x; i < V_DIM * (H_DIM / 8); i += 256) {
        int row = i >> 7;
        int c   = (i & 127) << 3;
        *reinterpret_cast<uint4*>(blds + row * LDB + c) =
            *reinterpret_cast<const uint4*>(woutT + (size_t)row * H_DIM + c);
    }
    __syncthreads();

    const int wave = threadIdx.x >> 5;
    const int lane = threadIdx.x & 31;
    const int m0   = (blockIdx.x * 8 + wave) * 16;   // 256*8 = 2048 M-tiles
    const int hi8  = (lane & 16) >> 1;

    const bf16_t* arow = hidden + (size_t)(m0 + (lane & 15)) * H_DIM
                                + (lane & 16);
    const bf16_t* bbase = blds + (lane & 15) * LDB + (lane & 16);

    v8f acc[8];
    const v8f vzero = {0.f, 0.f, 0.f, 0.f, 0.f, 0.f, 0.f, 0.f};
    #pragma unroll
    for (int nt = 0; nt < 8; ++nt) acc[nt] = vzero;

    for (int ks = 0; ks < H_DIM / 32; ++ks) {
        FragBF fa;
        const bf16_t* pa = arow + ks * 32;
        fa.q[0] = *reinterpret_cast<const uint4*>(pa);
        fa.q[1] = *reinterpret_cast<const uint4*>(pa + 8);
        #pragma unroll
        for (int nt = 0; nt < 8; ++nt) {
            FragBF fb;
            const bf16_t* pb = bbase + (size_t)(nt * 16) * LDB + ks * 32;
            fb.q[0] = *reinterpret_cast<const uint4*>(pb);
            fb.q[1] = *reinterpret_cast<const uint4*>(pb + 8);
            acc[nt] = __builtin_amdgcn_wmma_f32_16x16x32_bf16(
                          false, fa.v, false, fb.v, (short)0, acc[nt], false, false);
        }
    }

    #pragma unroll
    for (int nt = 0; nt < 8; ++nt) {
        int n = nt * 16 + (lane & 15);
        float bo = b_out[n];
        #pragma unroll
        for (int r = 0; r < 8; ++r) {
            int m = m0 + hi8 + r;
            logits[(size_t)m * V_DIM + n] = acc[nt][r] + bo;
        }
    }
}

// ---------------------------------------------------------------------------
// Host launcher
// ---------------------------------------------------------------------------
extern "C" void kernel_launch(void* const* d_in, const int* in_sizes, int n_in,
                              void* d_out, int out_size, void* d_ws, size_t ws_size,
                              hipStream_t stream) {
    const int*   x     = (const int*)  d_in[0];   // (64,512)
    const float* h0    = (const float*)d_in[1];   // (64,1024)
    const float* W_xh  = (const float*)d_in[2];   // (128,1024)
    const float* W_hh  = (const float*)d_in[3];   // (1024,1024)
    const float* b_h   = (const float*)d_in[4];   // (1024)
    const float* W_out = (const float*)d_in[5];   // (1024,128)
    const float* b_out = (const float*)d_in[6];   // (128)

    float* logits = (float*)d_out;                          // (64,512,128)
    float* hfinal = logits + (size_t)B_DIM * S_LEN * V_DIM; // (64,1024)

    // workspace layout (256B-aligned slabs)
    char* ws = (char*)d_ws;
    unsigned* bar    = (unsigned*)(ws + 0);                         // 256 B
    bf16_t*   hbuf   = (bf16_t*)(ws + 256);                         // 256 KB
    bf16_t*   whhT   = (bf16_t*)(ws + 256 + 262144);                // 2 MB
    bf16_t*   woutT  = (bf16_t*)(ws + 256 + 262144 + 2097152);      // 256 KB
    bf16_t*   hidden = (bf16_t*)(ws + 256 + 262144 + 2097152 + 262144); // 64 MB

    // init (barrier reset + h0 -> bf16) -- re-run every call for determinism
    rnn_init_kernel<<<(B_DIM * H_DIM + 255) / 256, 256, 0, stream>>>(h0, hbuf, bar);

    dim3 tb(32, 8);
    rnn_transpose_bf16_kernel<<<dim3(H_DIM / 32, H_DIM / 32), tb, 0, stream>>>(
        W_hh, whhT, H_DIM, H_DIM);
    rnn_transpose_bf16_kernel<<<dim3(V_DIM / 32, H_DIM / 32), tb, 0, stream>>>(
        W_out, woutT, H_DIM, V_DIM);

    size_t lds_rec  = (size_t)64 * (H_DIM + 8) * sizeof(bf16_t);    // 132 KB
    rnn_recurrence_kernel<<<NWG_REC, 256, lds_rec, stream>>>(
        x, W_xh, b_h, whhT, hbuf, hidden, hfinal, bar);

    size_t lds_proj = (size_t)V_DIM * (H_DIM + 8) * sizeof(bf16_t); // 258 KB
    rnn_logits_kernel<<<(B_DIM * S_LEN) / (8 * 16), 256, lds_proj, stream>>>(
        hidden, woutT, b_out, logits);
}
